// BiLSTMClass_58892591563432
// MI455X (gfx1250) — matrix operand order repaired and checked by reading, not compiled, not run on hardware
//
#include <hip/hip_runtime.h>

// ---------------------------------------------------------------------------
// BiLSTM (B=32, T=1024, I=H=1024) for MI455X / gfx1250, wave32 + WMMA bf16.
//
//   prologue (parallel): f32->bf16 convert of x and the 4 weight matrices,
//                        bias sums (bih+bhh), state init (c=c0, h=bf16(h0)).
//   1024 step kernels (sequential on stream): each launch advances BOTH
//     directions one timestep. 256 waves; each wave owns one 16x16 (batch x
//     hidden) tile and computes the i/f/g/o gate tiles with
//     v_wmma_f32_16x16x32_bf16 over K = I (x part) + H (h part).
//   The K loop is software-pipelined with two explicit fragment buffers so
//     next-iteration global loads overlap current WMMAs (no loadcnt-0 stall
//     per WMMA). Weights (32 MB bf16) stay resident in the 192 MB L2.
//   h is bf16 double-buffered across launches (RAW across waves); c is f32
//     in place (each element owned by exactly one wave). Final step writes
//     h directly to d_out as f32 (hf then hr).
// ---------------------------------------------------------------------------

#define B_ 32
#define T_ 1024
#define I_ 1024
#define H_ 1024

typedef __attribute__((ext_vector_type(16))) __bf16     v16bf;
typedef __attribute__((ext_vector_type(8)))  float      v8f;
typedef __attribute__((ext_vector_type(4)))  unsigned   v4u;

union ABFrag {
    v16bf v;
    v4u   q[2];
};

struct Frags {
    ABFrag a, b0, b1, b2, b3;
};

__device__ __forceinline__ unsigned short f32_to_bf16_rne(float f) {
    unsigned u = __float_as_uint(f);
    unsigned lsb = (u >> 16) & 1u;
    u += 0x7fffu + lsb;                 // round-to-nearest-even
    return (unsigned short)(u >> 16);
}

__device__ __forceinline__ float sigmoidf_(float x) {
    return 1.0f / (1.0f + __expf(-x));
}

// --------------------------- prologue kernels ------------------------------

__global__ void cvt_f32_bf16(const float* __restrict__ in,
                             unsigned short* __restrict__ out, int n) {
    int i = (blockIdx.x * blockDim.x + threadIdx.x) * 4;
    if (i + 3 < n) {
        float4 v = *(const float4*)(in + i);
        out[i + 0] = f32_to_bf16_rne(v.x);
        out[i + 1] = f32_to_bf16_rne(v.y);
        out[i + 2] = f32_to_bf16_rne(v.z);
        out[i + 3] = f32_to_bf16_rne(v.w);
    } else {
        for (int j = i; j < n; ++j) out[j] = f32_to_bf16_rne(in[j]);
    }
}

__global__ void bias_sum_kernel(const float* __restrict__ bihF,
                                const float* __restrict__ bhhF,
                                const float* __restrict__ bihR,
                                const float* __restrict__ bhhR,
                                float* __restrict__ bsum) {
    int j = blockIdx.x * blockDim.x + threadIdx.x;   // 0..4*H-1
    if (j < 4 * H_) {
        bsum[j]          = bihF[j] + bhhF[j];
        bsum[4 * H_ + j] = bihR[j] + bhhR[j];
    }
}

__global__ void init_state_kernel(const float* __restrict__ h0,
                                  const float* __restrict__ c0,
                                  float* __restrict__ c,                 // [2][B*H]
                                  unsigned short* __restrict__ h_par0) { // [2][B*H]
    int i = blockIdx.x * blockDim.x + threadIdx.x;   // 0..B*H-1
    if (i < B_ * H_) {
        float cv = c0[i];
        unsigned short hv = f32_to_bf16_rne(h0[i]);
        c[i]            = cv;
        c[B_ * H_ + i]  = cv;
        h_par0[i]           = hv;
        h_par0[B_ * H_ + i] = hv;
    }
}

// --------------------------- fragment helpers ------------------------------
// A fragment (16x32 bf16): lane<16: row=lane,     K = kb+{0..15}
//                          lane>=16: row=lane-16, K = kb+{16..31}
// B fragment (32x16 bf16): lane<16: col=lane,     K = kb+{0..15}
//                          lane>=16: col=lane-16, K = kb+{16..31}
//   B(k,n) = W[n][k]  (gates = act @ W^T), contiguous along k in W's row n.

__device__ __forceinline__ void ld5(Frags& f, const unsigned short* arow,
                                    const unsigned short* w0, const unsigned short* w1,
                                    const unsigned short* w2, const unsigned short* w3,
                                    int kb, int kh, int ko) {
    f.a.q[0]  = *(const v4u*)(arow + kb + kh);
    f.a.q[1]  = *(const v4u*)(arow + kb + kh + 8);
    f.b0.q[0] = *(const v4u*)(w0 + kb + ko);
    f.b0.q[1] = *(const v4u*)(w0 + kb + ko + 8);
    f.b1.q[0] = *(const v4u*)(w1 + kb + ko);
    f.b1.q[1] = *(const v4u*)(w1 + kb + ko + 8);
    f.b2.q[0] = *(const v4u*)(w2 + kb + ko);
    f.b2.q[1] = *(const v4u*)(w2 + kb + ko + 8);
    f.b3.q[0] = *(const v4u*)(w3 + kb + ko);
    f.b3.q[1] = *(const v4u*)(w3 + kb + ko + 8);
}

__device__ __forceinline__ void mm4(const Frags& f, v8f& ai, v8f& af, v8f& ag, v8f& ao) {
    ai = __builtin_amdgcn_wmma_f32_16x16x32_bf16(false, f.a.v, false, f.b0.v, (short)0, ai, false, false);
    af = __builtin_amdgcn_wmma_f32_16x16x32_bf16(false, f.a.v, false, f.b1.v, (short)0, af, false, false);
    ag = __builtin_amdgcn_wmma_f32_16x16x32_bf16(false, f.a.v, false, f.b2.v, (short)0, ag, false, false);
    ao = __builtin_amdgcn_wmma_f32_16x16x32_bf16(false, f.a.v, false, f.b3.v, (short)0, ao, false, false);
}

// Software-pipelined K=1024 GEMM phase: 4 gate tiles, double-buffered frags.
__device__ __forceinline__ void gemm_phase(const unsigned short* __restrict__ arow,
                                           const unsigned short* __restrict__ w0,
                                           const unsigned short* __restrict__ w1,
                                           const unsigned short* __restrict__ w2,
                                           const unsigned short* __restrict__ w3,
                                           int kh, int ko,
                                           v8f& ai, v8f& af, v8f& ag, v8f& ao) {
    Frags f0, f1;
    ld5(f0, arow, w0, w1, w2, w3, 0, kh, ko);
#pragma unroll 1
    for (int kb = 0; kb < I_ - 64; kb += 64) {
        ld5(f1, arow, w0, w1, w2, w3, kb + 32, kh, ko);   // issue before consuming f0
        mm4(f0, ai, af, ag, ao);
        ld5(f0, arow, w0, w1, w2, w3, kb + 64, kh, ko);   // issue before consuming f1
        mm4(f1, ai, af, ag, ao);
    }
    // kb == I_-64 handled here: fragments at I_-64 (in f0) and I_-32.
    ld5(f1, arow, w0, w1, w2, w3, I_ - 32, kh, ko);
    mm4(f0, ai, af, ag, ao);
    mm4(f1, ai, af, ag, ao);
}

// ----------------------------- step kernel ---------------------------------
// grid = 256 workgroups, block = 32 (one wave). wave -> (dir, mtile, ntile).
// C/D (16x16 f32, 8 VGPRs): VGPR r, lane l -> row = r + 8*(l>>4), col = l&15.

__global__ __launch_bounds__(32)
void lstm_step_kernel(const unsigned short* __restrict__ xbf,   // [B][T][I]
                      const unsigned short* __restrict__ wihF,  // [4H][I]
                      const unsigned short* __restrict__ whhF,  // [4H][H]
                      const unsigned short* __restrict__ wihR,
                      const unsigned short* __restrict__ whhR,
                      const float*          __restrict__ bsum,  // [2][4H]
                      float*                __restrict__ c,     // [2][B][H]
                      const unsigned short* __restrict__ hRd,   // [2][B][H] read parity
                      unsigned short*       __restrict__ hWr,   // [2][B][H] write parity
                      float*                __restrict__ hout,  // d_out on last step, else null
                      int s) {
    const int lane = threadIdx.x;
    const int w    = blockIdx.x;          // 0..255
    const int dir  = w >> 7;              // 0 fwd, 1 rev
    const int rem  = w & 127;
    const int mt   = rem & 1;             // batch tile (rows 0-15 / 16-31)
    const int nt   = rem >> 1;            // hidden tile 0..63
    const int t    = dir ? (T_ - 1 - s) : s;

    const unsigned short* Wih = dir ? wihR : wihF;
    const unsigned short* Whh = dir ? whhR : whhF;

    const int arow = lane & 15;           // A row within tile / B column (=W row)
    const int kh   = (lane >> 4) << 4;    // A: 0 or 16
    const int ko   = (lane >> 4) << 4;    // B: 0 or 16
    const int ncol = nt * 16 + arow;      // W row index within a gate block

    // Hoist bias + old-c loads so their latency hides under the GEMMs.
    const int n_local = lane & 15;
    const int m_base  = (lane >> 4) << 3;
    const int hcol    = nt * 16 + n_local;
    const float bi = bsum[dir * 4 * H_ + 0 * H_ + hcol];
    const float bf = bsum[dir * 4 * H_ + 1 * H_ + hcol];
    const float bg = bsum[dir * 4 * H_ + 2 * H_ + hcol];
    const float bo = bsum[dir * 4 * H_ + 3 * H_ + hcol];
    float cold[8];
#pragma unroll
    for (int r = 0; r < 8; ++r) {
        const int b = mt * 16 + m_base + r;
        cold[r] = c[(size_t)dir * B_ * H_ + (size_t)b * H_ + hcol];
    }

    v8f ai = {}, af = {}, ag = {}, ao = {};

    // ---- x_t @ Wih^T ----
    gemm_phase(xbf + ((size_t)(mt * 16 + arow) * T_ + t) * I_,
               Wih + (size_t)(0 * H_ + ncol) * I_,
               Wih + (size_t)(1 * H_ + ncol) * I_,
               Wih + (size_t)(2 * H_ + ncol) * I_,
               Wih + (size_t)(3 * H_ + ncol) * I_,
               kh, ko, ai, af, ag, ao);

    // ---- h_{t-1} @ Whh^T ----
    gemm_phase(hRd + (size_t)dir * B_ * H_ + (size_t)(mt * 16 + arow) * H_,
               Whh + (size_t)(0 * H_ + ncol) * H_,
               Whh + (size_t)(1 * H_ + ncol) * H_,
               Whh + (size_t)(2 * H_ + ncol) * H_,
               Whh + (size_t)(3 * H_ + ncol) * H_,
               kh, ko, ai, af, ag, ao);

    // ---- activations + state update (C layout: VGPR r, lane l) ----
#pragma unroll
    for (int r = 0; r < 8; ++r) {
        const int b = mt * 16 + m_base + r;
        const size_t idx = (size_t)dir * B_ * H_ + (size_t)b * H_ + hcol;
        const float si = sigmoidf_(ai[r] + bi);
        const float sf = sigmoidf_(af[r] + bf);
        const float tg = tanhf(ag[r] + bg);
        const float so = sigmoidf_(ao[r] + bo);
        const float cn = sf * cold[r] + si * tg;
        const float hn = so * tanhf(cn);
        c[idx]   = cn;
        hWr[idx] = f32_to_bf16_rne(hn);
        if (hout) hout[idx] = hn;       // last step: d_out = [hf | hr], f32
    }
}

// ------------------------------ launcher -----------------------------------

extern "C" void kernel_launch(void* const* d_in, const int* in_sizes, int n_in,
                              void* d_out, int out_size, void* d_ws, size_t ws_size,
                              hipStream_t stream) {
    (void)in_sizes; (void)n_in; (void)out_size; (void)ws_size;

    const float* x     = (const float*)d_in[0];   // [B,T,I]
    const float* h0    = (const float*)d_in[1];   // [B,H]
    const float* c0    = (const float*)d_in[2];   // [B,H]
    const float* Wih_f = (const float*)d_in[3];   // [4H,I]
    const float* Whh_f = (const float*)d_in[4];   // [4H,H]
    const float* bih_f = (const float*)d_in[5];   // [4H]
    const float* bhh_f = (const float*)d_in[6];
    const float* Wih_r = (const float*)d_in[7];
    const float* Whh_r = (const float*)d_in[8];
    const float* bih_r = (const float*)d_in[9];
    const float* bhh_r = (const float*)d_in[10];

    // ---- workspace carve (all re-written every call; no cross-call state) ----
    char* base = (char*)d_ws;
    auto carve = [&](size_t bytes) -> char* {
        char* r = base;
        base += (bytes + 255) & ~(size_t)255;
        return r;
    };
    unsigned short* xbf  = (unsigned short*)carve((size_t)B_ * T_ * I_ * 2);      // 64 MB
    unsigned short* wihF = (unsigned short*)carve((size_t)4 * H_ * I_ * 2);       // 8 MB
    unsigned short* whhF = (unsigned short*)carve((size_t)4 * H_ * H_ * 2);
    unsigned short* wihR = (unsigned short*)carve((size_t)4 * H_ * I_ * 2);
    unsigned short* whhR = (unsigned short*)carve((size_t)4 * H_ * H_ * 2);
    float*          bsum = (float*)carve((size_t)2 * 4 * H_ * 4);
    float*          cst  = (float*)carve((size_t)2 * B_ * H_ * 4);
    unsigned short* hbuf = (unsigned short*)carve((size_t)2 * 2 * B_ * H_ * 2);   // [parity][dir][B*H]

    // ---- prologue: conversions, bias sums, state init (all parallel) ----
    {
        int n = B_ * T_ * I_;
        cvt_f32_bf16<<<n / 1024, 256, 0, stream>>>(x, xbf, n);
    }
    {
        int n = 4 * H_ * I_;
        cvt_f32_bf16<<<n / 1024, 256, 0, stream>>>(Wih_f, wihF, n);
        cvt_f32_bf16<<<n / 1024, 256, 0, stream>>>(Whh_f, whhF, n);
        cvt_f32_bf16<<<n / 1024, 256, 0, stream>>>(Wih_r, wihR, n);
        cvt_f32_bf16<<<n / 1024, 256, 0, stream>>>(Whh_r, whhR, n);
    }
    bias_sum_kernel<<<(4 * H_ + 255) / 256, 256, 0, stream>>>(bih_f, bhh_f, bih_r, bhh_r, bsum);
    init_state_kernel<<<(B_ * H_ + 255) / 256, 256, 0, stream>>>(h0, c0, cst, hbuf /* parity 0 */);

    // ---- 1024 sequential timesteps; fwd & rev advance in the same launch ----
    const size_t par_stride = (size_t)2 * B_ * H_;   // elements per parity slab
    for (int s = 0; s < T_; ++s) {
        const int pr = s & 1;
        lstm_step_kernel<<<256, 32, 0, stream>>>(
            xbf, wihF, whhF, wihR, whhR, bsum, cst,
            hbuf + (size_t)pr * par_stride,
            hbuf + (size_t)(1 - pr) * par_stride,
            (s == T_ - 1) ? (float*)d_out : nullptr,
            s);
    }
}
